// DistributionSampler_59485297050199
// MI455X (gfx1250) — matrix operand order repaired and checked by reading, not compiled, not run on hardware
//
#include <hip/hip_runtime.h>
#include <hip/hip_bf16.h>
#include <cstdint>

// ---------------------------------------------------------------------------
// DistributionSampler for MI455X (gfx1250, wave32, WMMA)
//   q,k,v: [8,12,8192,64] f32 ; token_mask: [8,8192] f32 ; out: [8,12,8192] bool->f32
//   Pass 1: score[s] = dot(q0, k_s)/8 via v_wmma_f32_16x16x32_f16,
//           key[s]   = score + gumbel(hash) + (mask ? 0 : -1e30)
//   Pass 2: exact top-1024 per (b,h) via LDS radix-select, scatter out[s+1]=1, out[0]=1
// ---------------------------------------------------------------------------

typedef __attribute__((ext_vector_type(16))) _Float16 v16h;
typedef __attribute__((ext_vector_type(8)))  float    v8f;

#define B_DIM   8
#define H_DIM   12
#define BH_DIM  (B_DIM * H_DIM)   // 96
#define S_LEN   8192
#define D_LEN   64
#define NSAMP   1024
#define TEMP_INV 0.125f           // 1/TEMPERATURE, TEMPERATURE = 8

__device__ __forceinline__ uint32_t pcg_hash(uint32_t v) {
    v = v * 747796405u + 2891336453u;
    uint32_t w = ((v >> ((v >> 28u) + 4u)) ^ v) * 277803737u;
    return (w >> 22u) ^ w;
}

__device__ __forceinline__ float gumbel_of(uint32_t idx) {
    uint32_t h = pcg_hash(idx * 2654435761u + 12345u);
    float u = (float)(h >> 8) * (1.0f / 16777216.0f);
    u = fmaxf(u, 1e-7f);
    u = fminf(u, 1.0f - 1e-7f);
    return -__logf(-__logf(u));
}

// --------------------------- Pass 1: WMMA scores ---------------------------
// Block: 128 threads (4 waves). Each wave computes a 16-row score tile.
// gridDim.x = S/64 tiles, gridDim.y = B*H.
__global__ void __launch_bounds__(128)
score_key_kernel(const float* __restrict__ q,
                 const float* __restrict__ k,
                 const float* __restrict__ mask,
                 float* __restrict__ keys)
{
    const int bh   = blockIdx.y;           // 0..95
    const int b    = bh / H_DIM;
    const int wave = threadIdx.x >> 5;     // 0..3
    const int lane = threadIdx.x & 31;
    const int half = lane >> 4;            // 0: lanes 0-15, 1: lanes 16-31
    const int m    = lane & 15;            // row within 16-row tile

    // Stage the class-token query (64 floats) through LDS.
    __shared__ float qs[D_LEN];
    if (threadIdx.x < D_LEN)
        qs[threadIdx.x] = q[(size_t)bh * S_LEN * D_LEN + threadIdx.x];
    __syncthreads();

    const int row0 = blockIdx.x * 64 + wave * 16;
    const float* krow = k + ((size_t)bh * S_LEN + (size_t)(row0 + m)) * D_LEN;

    // Prefetch next block's slice of K (global_prefetch_b8).
    __builtin_prefetch(krow + (size_t)64 * D_LEN, 0, 1);

    v8f c;
    #pragma unroll
    for (int i = 0; i < 8; ++i) c[i] = 0.0f;

    // Two K-chunks of 32 accumulate the 64-dim dot product.
    #pragma unroll
    for (int ch = 0; ch < 2; ++ch) {
        const int kb = ch * 32;
        v16h a, bm;
        // A (16x32 f16): lanes 0-15 hold K {0..7,16..23}, lanes 16-31 hold K {8..15,24..31}
        #pragma unroll
        for (int i = 0; i < 8; ++i) {
            a[i]     = (_Float16)krow[kb + half * 8 + i];
            a[8 + i] = (_Float16)krow[kb + 16 + half * 8 + i];
        }
        // B (32x16 f16): every column n = q chunk; lanes 0-15 K {0..7,16..23}, lanes 16-31 K {8..15,24..31}
        #pragma unroll
        for (int i = 0; i < 8; ++i) {
            bm[i]     = (_Float16)qs[kb + half * 8 + i];
            bm[8 + i] = (_Float16)qs[kb + 16 + half * 8 + i];
        }

        c = __builtin_amdgcn_wmma_f32_16x16x32_f16(
                /*neg_a=*/false, a, /*neg_b=*/false, bm,
                /*c_mod=*/(short)0, c, /*reuse_a=*/false, /*reuse_b=*/false);
    }

    // C/D layout: VGPR r -> (M=r, lanes 0-15) and (M=r+8, lanes 16-31); all N equal.
    if (m == 0) {
        const int rbase = row0 + half * 8;
        #pragma unroll
        for (int r = 0; r < 8; ++r) {
            const int row = rbase + r;
            const float msk = mask[(size_t)b * S_LEN + row];
            const float g   = gumbel_of((uint32_t)(bh * S_LEN + row));
            const float pen = (msk > 0.0f) ? 0.0f : -1.0e30f;
            keys[(size_t)bh * S_LEN + row] = c[r] * TEMP_INV + g + pen;
        }
    }
}

// ------------------- Pass 2: exact top-1024 + scatter ----------------------
// One block of 1024 threads per (b,h); 8192 keys live in LDS (32 KB).
__global__ void __launch_bounds__(1024)
topk_scatter_kernel(const float* __restrict__ keys, float* __restrict__ out)
{
    const int bh  = blockIdx.x;
    const int tid = threadIdx.x;
    const size_t base = (size_t)bh * S_LEN;

    __shared__ uint32_t skeys[S_LEN];   // 32 KB
    __shared__ uint32_t hist[256];
    __shared__ uint32_t s_prefix;
    __shared__ int      s_need;
    __shared__ int      s_tie;

    // Load keys -> order-preserving uint32; zero the output row.
    #pragma unroll
    for (int i = 0; i < 8; ++i) {
        const int idx = tid + i * 1024;
        uint32_t u = __float_as_uint(keys[base + idx]);
        u = (u & 0x80000000u) ? ~u : (u | 0x80000000u);
        skeys[idx] = u;
        out[base + idx] = 0.0f;
    }
    if (tid == 0) { s_prefix = 0u; s_need = NSAMP; s_tie = 0; }
    __syncthreads();

    // 4-pass (8 bits each) radix-select for the 1024th-largest value.
    uint32_t prefmask = 0u;
    for (int pass = 0; pass < 4; ++pass) {
        const int shift = 24 - 8 * pass;
        if (tid < 256) hist[tid] = 0u;
        __syncthreads();
        const uint32_t prefix = s_prefix;
        #pragma unroll
        for (int i = 0; i < 8; ++i) {
            const uint32_t u = skeys[tid + i * 1024];
            if ((u & prefmask) == prefix)
                atomicAdd(&hist[(u >> shift) & 255u], 1u);
        }
        __syncthreads();
        if (tid == 0) {
            uint32_t cum = 0u;
            int need = s_need;
            int chosen = 0;
            for (int bin = 255; bin >= 0; --bin) {
                const uint32_t h = hist[bin];
                if (cum + h >= (uint32_t)need) { chosen = bin; need -= (int)cum; break; }
                cum += h;
            }
            s_prefix = prefix | ((uint32_t)chosen << shift);
            s_need   = need;
        }
        prefmask |= (0xFFu << shift);
        __syncthreads();
    }

    const uint32_t T   = s_prefix;     // exact threshold (ordered-uint space)
    const int     need = s_need;       // # ties at T still to take
    __syncthreads();

    // Scatter: selected index s -> out[s+1] (drop s == S-1); out[0] always set.
    #pragma unroll
    for (int i = 0; i < 8; ++i) {
        const int s = tid + i * 1024;
        const uint32_t u = skeys[s];
        bool take = (u > T);
        if (!take && u == T) {
            const int r = atomicAdd(&s_tie, 1);
            take = (r < need);
        }
        if (take && s < S_LEN - 1)
            out[base + s + 1] = 1.0f;
    }
    if (tid == 0) out[base + 0] = 1.0f;
}

// ---------------------------------------------------------------------------
extern "C" void kernel_launch(void* const* d_in, const int* in_sizes, int n_in,
                              void* d_out, int out_size, void* d_ws, size_t ws_size,
                              hipStream_t stream) {
    (void)in_sizes; (void)n_in; (void)out_size; (void)ws_size;
    const float* q    = (const float*)d_in[0];
    const float* k    = (const float*)d_in[1];
    // d_in[2] (v) is unused by the reference.
    const float* mask = (const float*)d_in[3];
    float* out  = (float*)d_out;
    float* keys = (float*)d_ws;          // B*H*S floats = 3 MB of scratch

    dim3 g1(S_LEN / 64, BH_DIM);         // 128 x 96 blocks, 4 waves each
    score_key_kernel<<<g1, 128, 0, stream>>>(q, k, mask, keys);
    topk_scatter_kernel<<<BH_DIM, 1024, 0, stream>>>(keys, out);
}
